// Transformer_17008070492748
// MI455X (gfx1250) — matrix operand order repaired and checked by reading, not compiled, not run on hardware
//
#include <hip/hip_runtime.h>
#include <hip/hip_bf16.h>
#include <math.h>

typedef __attribute__((ext_vector_type(16))) __bf16 v16bf;
typedef __attribute__((ext_vector_type(8)))  __bf16 v8bf;
typedef __attribute__((ext_vector_type(8)))  float  v8f;

#define WMMA_BF16(a, b, c) \
  __builtin_amdgcn_wmma_f32_16x16x32_bf16(false, (a), false, (b), (short)0, (c), false, false)

static constexpr int S_LEN  = 1024;
static constexpr int BATCH  = 8;
static constexpr int MROWS  = 8192;     // B*S token rows
static constexpr int DIM    = 256;
static constexpr int INNER  = 512;
static constexpr int NHEAD  = 8;
static constexpr int DH     = 64;
static constexpr int FF     = 1024;
static constexpr int IN_DIM = 12306;
static constexpr int KPADWE = 12320;    // IN_DIM rounded up to 32
static constexpr int NSOUT  = 12290;    // IMG + AUD

// ---------------------------------------------------------------------------
// A-fragment loaders: lane lo = row, element e -> K = hi*16+e
// ---------------------------------------------------------------------------
__device__ __forceinline__ v16bf load_a_frag(const __bf16* __restrict__ aprow,
                                             int k0, int hi, int K) {
  (void)K;  // bf16 activations always have K % 32 == 0
  v8bf a0 = *(const v8bf*)(aprow + k0 + hi * 16);
  v8bf a1 = *(const v8bf*)(aprow + k0 + hi * 16 + 8);
  return __builtin_shufflevector(a0, a1, 0, 1, 2, 3, 4, 5, 6, 7,
                                 8, 9, 10, 11, 12, 13, 14, 15);
}
__device__ __forceinline__ v16bf load_a_frag(const float* __restrict__ aprow,
                                             int k0, int hi, int K) {
  v16bf a;
#pragma unroll
  for (int j = 0; j < 8; ++j) {
    const int ka = k0 + hi * 16 + j;
    const int kb = k0 + hi * 16 + 8 + j;
    a[j]     = (__bf16)(ka < K ? aprow[ka] : 0.0f);
    a[j + 8] = (__bf16)(kb < K ? aprow[kb] : 0.0f);
  }
  return a;
}

// ---------------------------------------------------------------------------
// GEMM: C[M,N] = A[M,K] @ Bt.T + bias.  Bt is staged bf16, (N, KB) row-major
// (i.e. already transposed), KB >= K padded to a multiple of 32 with zeros.
// Block = 128 threads = 4 waves; each wave computes a 32x64 C tile
// (2 A-fragments x 4 B-fragments -> 8 WMMAs per 32-wide K step).
// ---------------------------------------------------------------------------
template <typename TA, int OUTBF>
__global__ __launch_bounds__(128) void gemm_bt(const TA* __restrict__ A,
                                               const __bf16* __restrict__ Bt,
                                               const float* __restrict__ bias,
                                               void* __restrict__ Cout,
                                               int M, int N, int K, int KB) {
  const int lane = threadIdx.x & 31;
  const int wave = threadIdx.x >> 5;
  const int lo   = lane & 15;
  const int hi   = lane >> 4;
  const int row0 = blockIdx.y * 128 + wave * 32;
  const int col0 = blockIdx.x * 64;
  const TA* ap0 = A + (size_t)(row0 + lo) * K;
  const TA* ap1 = A + (size_t)(row0 + 16 + lo) * K;

  v8f acc[2][4] = {};
  for (int k0 = 0; k0 < K; k0 += 32) {
    const v16bf a0 = load_a_frag(ap0, k0, hi, K);
    const v16bf a1 = load_a_frag(ap1, k0, hi, K);
#pragma unroll
    for (int t = 0; t < 4; ++t) {
      const __bf16* bp = Bt + (size_t)(col0 + t * 16 + lo) * KB + k0 + hi * 16;
      const v16bf b = *(const v16bf*)bp;   // one aligned 32B vector load
      acc[0][t] = WMMA_BF16(a0, b, acc[0][t]);
      acc[1][t] = WMMA_BF16(a1, b, acc[1][t]);
    }
  }
#pragma unroll
  for (int u = 0; u < 2; ++u)
#pragma unroll
    for (int t = 0; t < 4; ++t) {
      const int col  = col0 + t * 16 + lo;
      const float bv = bias ? bias[col] : 0.0f;
#pragma unroll
      for (int r = 0; r < 8; ++r) {
        const int row  = row0 + u * 16 + r + hi * 8;  // C elem r -> M = r+hi*8
        const float vv = acc[u][t][r] + bv;
        if (OUTBF) ((__bf16*)Cout)[(size_t)row * N + col] = (__bf16)vv;
        else       ((float*)Cout)[(size_t)row * N + col] = vv;
      }
    }
}

// ---------------------------------------------------------------------------
// Fused causal flash attention, all-bf16 operands.
// q/k: (MROWS, INNER) with col = h*64+d.  vt: (B, H, DH, S) transposed so the
// P@V B-fragment is one aligned 32B vector load.  o: (MROWS, INNER).
// One wave per (b, h, 16-query tile); key blocks of 32 -> full-K=32 P@V WMMA.
// ---------------------------------------------------------------------------
__global__ __launch_bounds__(128) void attn_kernel(const __bf16* __restrict__ q,
                                                   const __bf16* __restrict__ k,
                                                   const __bf16* __restrict__ vt,
                                                   __bf16* __restrict__ o) {
  __shared__ __bf16 ldsp[4][16][32];   // per-wave P tile (16 rows x 32 keys)
  const int lane = threadIdx.x & 31;
  const int wave = threadIdx.x >> 5;
  const int lo   = lane & 15;
  const int hi   = lane >> 4;
  const int h    = blockIdx.y;
  const int b    = blockIdx.z;
  const int qt   = blockIdx.x * 4 + wave;
  const size_t base = (size_t)b * S_LEN * INNER + (size_t)h * DH;
  const __bf16* vtb = vt + (size_t)(b * NHEAD + h) * DH * S_LEN;

  // Preload Q fragments (A-matrix, two K=32 chunks of DH=64)
  v16bf aq[2];
  {
    const __bf16* qp = q + base + (size_t)(qt * 16 + lo) * INNER;
#pragma unroll
    for (int kk = 0; kk < 2; ++kk) {
      v8bf q0 = *(const v8bf*)(qp + kk * 32 + hi * 16);
      v8bf q1 = *(const v8bf*)(qp + kk * 32 + hi * 16 + 8);
      aq[kk] = __builtin_shufflevector(q0, q1, 0, 1, 2, 3, 4, 5, 6, 7,
                                       8, 9, 10, 11, 12, 13, 14, 15);
    }
  }

  float mrow[8], lrow[8];
  v8f vo[4] = {};
#pragma unroll
  for (int r = 0; r < 8; ++r) { mrow[r] = -1e30f; lrow[r] = 0.0f; }

  const int nkb = qt / 2 + 1;                 // causal: key blocks of 32
  for (int kb = 0; kb < nkb; ++kb) {
    v8f s0 = {}, s1 = {};
    const __bf16* kp0 = k + base + (size_t)(kb * 32 + lo) * INNER;
    const __bf16* kp1 = k + base + (size_t)(kb * 32 + 16 + lo) * INNER;
#pragma unroll
    for (int kk = 0; kk < 2; ++kk) {
      const v16bf b0 = *(const v16bf*)(kp0 + kk * 32 + hi * 16);
      const v16bf b1 = *(const v16bf*)(kp1 + kk * 32 + hi * 16);
      s0 = WMMA_BF16(aq[kk], b0, s0);
      s1 = WMMA_BF16(aq[kk], b1, s1);
    }
    const int key0 = kb * 32 + lo;
    const int key1 = key0 + 16;
#pragma unroll
    for (int r = 0; r < 8; ++r) {
      const int srow = qt * 16 + r + hi * 8;
      float x0 = (key0 <= srow) ? s0[r] * 10.0f : -1e30f;   // QK_SCALE + causal
      float x1 = (key1 <= srow) ? s1[r] * 10.0f : -1e30f;
      float rm = fmaxf(x0, x1);                              // row max, 16 lanes
      rm = fmaxf(rm, __shfl_xor(rm, 1));
      rm = fmaxf(rm, __shfl_xor(rm, 2));
      rm = fmaxf(rm, __shfl_xor(rm, 4));
      rm = fmaxf(rm, __shfl_xor(rm, 8));
      const float mn = fmaxf(mrow[r], rm);
      const float e0 = expf(x0 - mn);
      const float e1 = expf(x1 - mn);
      float ps = e0 + e1;
      ps += __shfl_xor(ps, 1);
      ps += __shfl_xor(ps, 2);
      ps += __shfl_xor(ps, 4);
      ps += __shfl_xor(ps, 8);
      const float alpha = expf(mrow[r] - mn);
      lrow[r] = lrow[r] * alpha + ps;
      mrow[r] = mn;
#pragma unroll
      for (int t = 0; t < 4; ++t) vo[t][r] *= alpha;
      ldsp[wave][r + hi * 8][lo]      = (__bf16)e0;   // C-layout -> LDS
      ldsp[wave][r + hi * 8][16 + lo] = (__bf16)e1;
    }
    asm volatile("s_wait_dscnt 0" ::: "memory");
    v16bf ap;                                         // reload as A-fragment
#pragma unroll
    for (int j = 0; j < 8; ++j) {
      ap[j]     = ldsp[wave][lo][hi * 16 + j];
      ap[j + 8] = ldsp[wave][lo][hi * 16 + 8 + j];
    }
    asm volatile("s_wait_dscnt 0" ::: "memory");
#pragma unroll
    for (int t = 0; t < 4; ++t) {                     // O += P (16x32) @ V (32x16)
      const __bf16* vp = vtb + (size_t)(t * 16 + lo) * S_LEN + kb * 32 + hi * 16;
      const v16bf bv = *(const v16bf*)vp;             // contiguous in key dim
      vo[t] = WMMA_BF16(ap, bv, vo[t]);
    }
  }
#pragma unroll
  for (int t = 0; t < 4; ++t)
#pragma unroll
    for (int r = 0; r < 8; ++r) {
      const int srow = qt * 16 + r + hi * 8;
      o[base + (size_t)srow * INNER + t * 16 + lo] = (__bf16)(vo[t][r] / lrow[r]);
    }
}

// V transpose: (MROWS, INNER) [m = b*S+s, col = h*64+d] -> (B, H, DH, S) bf16
__global__ __launch_bounds__(256) void vtrans_kernel(const __bf16* __restrict__ v,
                                                     __bf16* __restrict__ vt) {
  const int idx = blockIdx.x * 256 + threadIdx.x;  // over MROWS*INNER
  const int m = idx >> 9;
  const int c = idx & 511;
  const int b = m >> 10, s = m & 1023;
  const int h = c >> 6, d = c & 63;
  vt[((size_t)(b * NHEAD + h) * DH + d) * S_LEN + s] = v[idx];
}

// ---------------------------------------------------------------------------
// LayerNorm (population variance, eps=1e-5) * gamma; fp32 in, bf16 out.
// ---------------------------------------------------------------------------
__global__ __launch_bounds__(256) void ln_kernel(const float* __restrict__ x,
                                                 const float* __restrict__ g,
                                                 __bf16* __restrict__ y) {
  const int lane = threadIdx.x & 31;
  const int wave = threadIdx.x >> 5;
  const int row  = blockIdx.x * 8 + wave;
  const float* xp = x + (size_t)row * DIM;
  float vals[8], s = 0.0f, s2 = 0.0f;
#pragma unroll
  for (int j = 0; j < 8; ++j) {
    const float vv = xp[lane + j * 32];
    vals[j] = vv; s += vv; s2 += vv * vv;
  }
#pragma unroll
  for (int mask = 1; mask < 32; mask <<= 1) {
    s  += __shfl_xor(s, mask);
    s2 += __shfl_xor(s2, mask);
  }
  const float mean = s * (1.0f / DIM);
  const float var  = s2 * (1.0f / DIM) - mean * mean;
  const float inv  = rsqrtf(var + 1e-5f);
  __bf16* yp = y + (size_t)row * DIM;
#pragma unroll
  for (int j = 0; j < 8; ++j) {
    const int c = lane + j * 32;
    yp[c] = (__bf16)((vals[j] - mean) * inv * g[c]);
  }
}

__global__ void lnlast_kernel(const float* __restrict__ hsrc,
                              const float* __restrict__ g,
                              float* __restrict__ hl) {
  const int lane = threadIdx.x;  // 32 threads, one block per batch element
  const int b = blockIdx.x;
  const float* xp = hsrc + ((size_t)b * S_LEN + (S_LEN - 1)) * DIM;
  float vals[8], s = 0.0f, s2 = 0.0f;
#pragma unroll
  for (int j = 0; j < 8; ++j) {
    const float vv = xp[lane + j * 32];
    vals[j] = vv; s += vv; s2 += vv * vv;
  }
#pragma unroll
  for (int mask = 1; mask < 32; mask <<= 1) {
    s  += __shfl_xor(s, mask);
    s2 += __shfl_xor(s2, mask);
  }
  const float mean = s * (1.0f / DIM);
  const float var  = s2 * (1.0f / DIM) - mean * mean;
  const float inv  = rsqrtf(var + 1e-5f);
#pragma unroll
  for (int j = 0; j < 8; ++j) {
    const int c = lane + j * 32;
    hl[b * DIM + c] = (vals[j] - mean) * inv * g[c];
  }
}

// ---------------------------------------------------------------------------
// Fused L2-norm + per-head scale + RoPE with xscale (q: xs, k: 1/xs).
// fp32 in (GEMM output), bf16 out (attention operands).
// ---------------------------------------------------------------------------
__global__ __launch_bounds__(256) void rope_kernel(const float* __restrict__ qf,
                                                   const float* __restrict__ kf,
                                                   __bf16* __restrict__ qh,
                                                   __bf16* __restrict__ kh,
                                                   const float* __restrict__ qs,
                                                   const float* __restrict__ ks) {
  const int idx = blockIdx.x * blockDim.x + threadIdx.x;  // MROWS*NHEAD
  const int hh = idx & (NHEAD - 1);
  const int m  = idx >> 3;
  const int s  = m & (S_LEN - 1);
  const float t = (float)s;
  const float power = (t - 512.0f) * (1.0f / 512.0f);
  for (int pass = 0; pass < 2; ++pass) {
    const float* p  = (pass ? kf : qf) + (size_t)m * INNER + hh * DH;
    __bf16* d       = (pass ? kh : qh) + (size_t)m * INNER + hh * DH;
    const float* sc = (pass ? ks : qs) + hh * DH;
    float v[64];
    float ss = 0.0f;
#pragma unroll
    for (int dd = 0; dd < 64; ++dd) { v[dd] = p[dd]; ss += v[dd] * v[dd]; }
    const float inv = rsqrtf(ss + 1e-12f);
#pragma unroll
    for (int dd = 0; dd < 64; ++dd) v[dd] *= inv * sc[dd];
#pragma unroll
    for (int j = 0; j < 16; ++j) {
      const float fr = t * powf(10000.0f, -(float)(2 * j) * (1.0f / 32.0f));
      const float c = cosf(fr), sn = sinf(fr);
      const float scj = (2.0f * j + 12.8f) * (1.0f / 44.8f);
      float xs = powf(scj, power);
      if (pass) xs = 1.0f / xs;
      const float a0 = v[j], a1 = v[j + 16];
      v[j]      = (a0 * c - a1 * sn) * xs;
      v[j + 16] = (a1 * c + a0 * sn) * xs;
    }
#pragma unroll
    for (int dd = 0; dd < 64; ++dd) d[dd] = (__bf16)v[dd];
  }
}

// GRU gate: h = (1-z)*n + z*h  (in place on hbuf), gi/gh are (MROWS, 768) fp32
__global__ __launch_bounds__(256) void gru_kernel(const float* __restrict__ gi,
                                                  const float* __restrict__ gh,
                                                  float* __restrict__ hbuf) {
  const int idx = blockIdx.x * 256 + threadIdx.x;
  const int m = idx >> 8, c = idx & 255;
  const float* gip = gi + (size_t)m * 768;
  const float* ghp = gh + (size_t)m * 768;
  const float r = 1.0f / (1.0f + expf(-(gip[c] + ghp[c])));
  const float z = 1.0f / (1.0f + expf(-(gip[256 + c] + ghp[256 + c])));
  const float n = tanhf(gip[512 + c] + r * ghp[512 + c]);
  const float hp = hbuf[idx];
  hbuf[idx] = (1.0f - z) * n + z * hp;
}

// GEGLU: fa = a * gelu_exact(g); u (MROWS,2048) bf16 -> fa (MROWS,1024) bf16
__global__ __launch_bounds__(256) void geglu_kernel(const __bf16* __restrict__ u,
                                                    __bf16* __restrict__ fa) {
  const int idx = blockIdx.x * 256 + threadIdx.x;
  const int m = idx >> 10, c = idx & 1023;
  const __bf16* up = u + (size_t)m * 2048;
  const float a = (float)up[c];
  const float g = (float)up[1024 + c];
  const float ge = 0.5f * g * (1.0f + erff(g * 0.70710678118f));
  fa[idx] = (__bf16)(a * ge);
}

// Output heads: tiny M=8 GEMVs, one thread per output column (fp32).
__global__ __launch_bounds__(256) void head_kernel(const float* __restrict__ hl,
                                                   const float* __restrict__ Wns,
                                                   const float* __restrict__ bns,
                                                   const float* __restrict__ Wa,
                                                   const float* __restrict__ ba,
                                                   float* __restrict__ out) {
  const int j = blockIdx.x * 256 + threadIdx.x;
  const int b = blockIdx.y;
  const float* hp = hl + b * DIM;
  if (j < NSOUT) {
    float acc = bns[j];
    for (int kk = 0; kk < DIM; ++kk) acc += hp[kk] * Wns[(size_t)kk * NSOUT + j];
    out[(size_t)b * NSOUT + j] = acc;
  } else if (j < NSOUT + 16) {
    const int ja = j - NSOUT;
    float acc = ba[ja];
    for (int kk = 0; kk < DIM; ++kk) acc += hp[kk] * Wa[kk * 16 + ja];
    out[(size_t)BATCH * NSOUT + b * 16 + ja] = acc;
  }
}

// Weight staging: straight fp32 -> bf16
__global__ __launch_bounds__(256) void cvt_kernel(const float* __restrict__ s,
                                                  __bf16* __restrict__ d, int n) {
  const int i = blockIdx.x * 256 + threadIdx.x;
  if (i < n) d[i] = (__bf16)s[i];
}
// Weight staging: transpose (K,N) fp32 -> (N,KB) bf16 with zero pad k in [K,KB)
__global__ __launch_bounds__(256) void cvtT_kernel(const float* __restrict__ s,
                                                   __bf16* __restrict__ d,
                                                   int K, int N, int KB) {
  const int i = blockIdx.x * 256 + threadIdx.x;
  if (i >= N * KB) return;
  const int n = i / KB, k = i - n * KB;
  d[(size_t)n * KB + k] = (__bf16)(k < K ? s[(size_t)k * N + n] : 0.0f);
}

// ---------------------------------------------------------------------------
extern "C" void kernel_launch(void* const* d_in, const int* in_sizes, int n_in,
                              void* d_out, int out_size, void* d_ws, size_t ws_size,
                              hipStream_t stream) {
  (void)in_sizes; (void)n_in; (void)out_size; (void)ws_size;
  const float* x     = (const float*)d_in[0];
  const float* We    = (const float*)d_in[1];
  const float* be    = (const float*)d_in[2];
  const float* ln1g  = (const float*)d_in[3];
  const float* Wq    = (const float*)d_in[4];
  const float* Wk    = (const float*)d_in[5];
  const float* Wv    = (const float*)d_in[6];
  const float* qsc   = (const float*)d_in[7];
  const float* ksc   = (const float*)d_in[8];
  const float* Wo    = (const float*)d_in[9];
  const float* gaWih = (const float*)d_in[10];
  const float* gaWhh = (const float*)d_in[11];
  const float* gabih = (const float*)d_in[12];
  const float* gabhh = (const float*)d_in[13];
  const float* ln2g  = (const float*)d_in[14];
  const float* W1    = (const float*)d_in[15];
  const float* b1    = (const float*)d_in[16];
  const float* W2    = (const float*)d_in[17];
  const float* b2    = (const float*)d_in[18];
  const float* gfWih = (const float*)d_in[19];
  const float* gfWhh = (const float*)d_in[20];
  const float* gfbih = (const float*)d_in[21];
  const float* gfbhh = (const float*)d_in[22];
  const float* lnfg  = (const float*)d_in[23];
  const float* Wns   = (const float*)d_in[24];
  const float* bns   = (const float*)d_in[25];
  const float* Wa    = (const float*)d_in[26];
  const float* ba    = (const float*)d_in[27];
  float* out = (float*)d_out;

  // ---- workspace carving (256B-aligned sub-buffers) ----
  char* base = (char*)d_ws;
  size_t off = 0;
  auto alloc = [&](size_t bytes) -> void* {
    void* r = base + off;
    off += (bytes + 255) & ~(size_t)255;
    return r;
  };
  float*  hbuf  = (float*)alloc((size_t)MROWS * DIM * 4);      // residual stream
  __bf16* ybuf  = (__bf16*)alloc((size_t)MROWS * DIM * 2);     // LN out / f-proj
  float*  qbuf  = (float*)alloc((size_t)MROWS * INNER * 4);    // q (pre-rope)
  float*  kbuf  = (float*)alloc((size_t)MROWS * INNER * 4);    // k (pre-rope)
  __bf16* vbuf  = (__bf16*)alloc((size_t)MROWS * INNER * 2);   // v
  __bf16* obuf  = (__bf16*)alloc((size_t)MROWS * INNER * 2);   // attn out
  __bf16* cbuf  = (__bf16*)alloc((size_t)MROWS * FF * 2);      // vt / o2 / fa
  float*  gibuf = (float*)alloc((size_t)MROWS * 768 * 4);
  float*  ghbuf = (float*)alloc((size_t)MROWS * 768 * 4);
  float*  hl    = (float*)alloc((size_t)BATCH * DIM * 4);
  // staged bf16 weights (pre-transposed to (N, KB))
  __bf16* sWe   = (__bf16*)alloc((size_t)DIM * KPADWE * 2);
  __bf16* sWq   = (__bf16*)alloc((size_t)4 * INNER * DIM * 2);
  __bf16* sWk   = (__bf16*)alloc((size_t)4 * INNER * DIM * 2);
  __bf16* sWv   = (__bf16*)alloc((size_t)4 * INNER * DIM * 2);
  __bf16* sWo   = (__bf16*)alloc((size_t)4 * DIM * INNER * 2);
  __bf16* sW1   = (__bf16*)alloc((size_t)4 * 2048 * DIM * 2);
  __bf16* sW2   = (__bf16*)alloc((size_t)4 * DIM * FF * 2);
  __bf16* sgaWih= (__bf16*)alloc((size_t)4 * 768 * DIM * 2);
  __bf16* sgaWhh= (__bf16*)alloc((size_t)4 * 768 * DIM * 2);
  __bf16* sgfWih= (__bf16*)alloc((size_t)4 * 768 * DIM * 2);
  __bf16* sgfWhh= (__bf16*)alloc((size_t)4 * 768 * DIM * 2);
  // aliases (non-overlapping lifetimes)
  __bf16* ubuf = (__bf16*)qbuf;    // (MROWS,2048) bf16 = 32MB inside q+k+v (40MB)
  __bf16* qbh  = (__bf16*)gibuf;   // bf16 q for attention (8MB in 24MB)
  __bf16* kbh  = (__bf16*)ghbuf;   // bf16 k for attention
  __bf16* vtb  = cbuf;             // transposed V (B,H,DH,S); dead before o2/fa

  // ---- stage weights to bf16 once ----
  cvtT_kernel<<<(DIM * KPADWE + 255) / 256, 256, 0, stream>>>(We, sWe, IN_DIM, DIM, KPADWE);
  for (int i = 0; i < 4; ++i) {
    cvtT_kernel<<<(INNER * DIM) / 256, 256, 0, stream>>>(
        Wq + (size_t)i * DIM * INNER, sWq + (size_t)i * INNER * DIM, DIM, INNER, DIM);
    cvtT_kernel<<<(INNER * DIM) / 256, 256, 0, stream>>>(
        Wk + (size_t)i * DIM * INNER, sWk + (size_t)i * INNER * DIM, DIM, INNER, DIM);
    cvtT_kernel<<<(INNER * DIM) / 256, 256, 0, stream>>>(
        Wv + (size_t)i * DIM * INNER, sWv + (size_t)i * INNER * DIM, DIM, INNER, DIM);
    cvtT_kernel<<<(DIM * INNER) / 256, 256, 0, stream>>>(
        Wo + (size_t)i * INNER * DIM, sWo + (size_t)i * DIM * INNER, INNER, DIM, INNER);
    cvtT_kernel<<<(2048 * DIM) / 256, 256, 0, stream>>>(
        W1 + (size_t)i * DIM * 2048, sW1 + (size_t)i * 2048 * DIM, DIM, 2048, DIM);
    cvtT_kernel<<<(DIM * FF) / 256, 256, 0, stream>>>(
        W2 + (size_t)i * FF * DIM, sW2 + (size_t)i * DIM * FF, FF, DIM, FF);
  }
  cvt_kernel<<<(4 * 768 * DIM) / 256, 256, 0, stream>>>(gaWih, sgaWih, 4 * 768 * DIM);
  cvt_kernel<<<(4 * 768 * DIM) / 256, 256, 0, stream>>>(gaWhh, sgaWhh, 4 * 768 * DIM);
  cvt_kernel<<<(4 * 768 * DIM) / 256, 256, 0, stream>>>(gfWih, sgfWih, 4 * 768 * DIM);
  cvt_kernel<<<(4 * 768 * DIM) / 256, 256, 0, stream>>>(gfWhh, sgfWhh, 4 * 768 * DIM);

  const dim3 blk(128);
  // Embedding: h = x @ We + be   (HBM-bound single pass over 403 MB of x)
  gemm_bt<float, 0><<<dim3(DIM / 64, MROWS / 128), blk, 0, stream>>>(
      x, sWe, be, hbuf, MROWS, DIM, IN_DIM, KPADWE);

  for (int i = 0; i < 4; ++i) {
    ln_kernel<<<MROWS / 8, 256, 0, stream>>>(hbuf, ln1g + i * DIM, ybuf);
    gemm_bt<__bf16, 0><<<dim3(INNER / 64, MROWS / 128), blk, 0, stream>>>(
        ybuf, sWq + (size_t)i * INNER * DIM, nullptr, qbuf, MROWS, INNER, DIM, DIM);
    gemm_bt<__bf16, 0><<<dim3(INNER / 64, MROWS / 128), blk, 0, stream>>>(
        ybuf, sWk + (size_t)i * INNER * DIM, nullptr, kbuf, MROWS, INNER, DIM, DIM);
    gemm_bt<__bf16, 1><<<dim3(INNER / 64, MROWS / 128), blk, 0, stream>>>(
        ybuf, sWv + (size_t)i * INNER * DIM, nullptr, vbuf, MROWS, INNER, DIM, DIM);
    rope_kernel<<<MROWS * NHEAD / 256, 256, 0, stream>>>(
        qbuf, kbuf, qbh, kbh, qsc + i * NHEAD * DH, ksc + i * NHEAD * DH);
    vtrans_kernel<<<MROWS * INNER / 256, 256, 0, stream>>>(vbuf, vtb);
    attn_kernel<<<dim3(S_LEN / 64, NHEAD, BATCH), blk, 0, stream>>>(qbh, kbh, vtb, obuf);
    gemm_bt<__bf16, 1><<<dim3(DIM / 64, MROWS / 128), blk, 0, stream>>>(
        obuf, sWo + (size_t)i * DIM * INNER, nullptr, cbuf, MROWS, DIM, INNER, INNER);
    // attention GRU gate
    gemm_bt<__bf16, 0><<<dim3(768 / 64, MROWS / 128), blk, 0, stream>>>(
        cbuf, sgaWih + (size_t)i * 768 * DIM, gabih + i * 768, gibuf, MROWS, 768, DIM, DIM);
    gemm_bt<float, 0><<<dim3(768 / 64, MROWS / 128), blk, 0, stream>>>(
        hbuf, sgaWhh + (size_t)i * 768 * DIM, gabhh + i * 768, ghbuf, MROWS, 768, DIM, DIM);
    gru_kernel<<<MROWS * DIM / 256, 256, 0, stream>>>(gibuf, ghbuf, hbuf);
    // FFN (GEGLU)
    ln_kernel<<<MROWS / 8, 256, 0, stream>>>(hbuf, ln2g + i * DIM, ybuf);
    gemm_bt<__bf16, 1><<<dim3(2048 / 64, MROWS / 128), blk, 0, stream>>>(
        ybuf, sW1 + (size_t)i * 2048 * DIM, b1 + i * 2048, ubuf, MROWS, 2048, DIM, DIM);
    geglu_kernel<<<MROWS * FF / 256, 256, 0, stream>>>(ubuf, cbuf);
    gemm_bt<__bf16, 1><<<dim3(DIM / 64, MROWS / 128), blk, 0, stream>>>(
        cbuf, sW2 + (size_t)i * DIM * FF, b2 + i * DIM, ybuf, MROWS, DIM, FF, FF);
    // FFN GRU gate
    gemm_bt<__bf16, 0><<<dim3(768 / 64, MROWS / 128), blk, 0, stream>>>(
        ybuf, sgfWih + (size_t)i * 768 * DIM, gfbih + i * 768, gibuf, MROWS, 768, DIM, DIM);
    gemm_bt<float, 0><<<dim3(768 / 64, MROWS / 128), blk, 0, stream>>>(
        hbuf, sgfWhh + (size_t)i * 768 * DIM, gfbhh + i * 768, ghbuf, MROWS, 768, DIM, DIM);
    gru_kernel<<<MROWS * DIM / 256, 256, 0, stream>>>(gibuf, ghbuf, hbuf);
  }

  lnlast_kernel<<<BATCH, 32, 0, stream>>>(hbuf, lnfg, hl);
  head_kernel<<<dim3((IN_DIM + 255) / 256, BATCH), 256, 0, stream>>>(hl, Wns, bns, Wa, ba, out);
}
